// IPM_67834713473340
// MI455X (gfx1250) — matrix operand order repaired and checked by reading, not compiled, not run on hardware
//
#include <hip/hip_runtime.h>
#include <hip/hip_bf16.h>

typedef __attribute__((ext_vector_type(16))) __bf16          v16bf;
typedef __attribute__((ext_vector_type(16))) unsigned short  v16u;
typedef __attribute__((ext_vector_type(8)))  unsigned short  v8u;
typedef __attribute__((ext_vector_type(8)))  float           v8f;

#define BEV_W   240
#define BEV_H   120
#define NLVL    5
#define NPIX    (BEV_H * BEV_W)        // 28800
#define NCAM    6
#define CIN     128
#define CFEAT   655                    // (128+3)*5 input channels to conv
#define KC      672                    // channels padded to 21*32
#define NKB     21                     // K-blocks of 32 per tap
#define NSTEP   (9 * NKB)              // 189 WMMA steps per tile
#define COUT    128
#define IMH     64
#define IMW     176
#define TAPB    (NKB * 512)            // 10752 ushorts = 21504 B of B per tap
#define STAGEB  24576                  // staged bytes per tap (256 thr * 6 * 16B)

__device__ __forceinline__ unsigned short f2bf(float f) {
    unsigned int u = __builtin_bit_cast(unsigned int, f);
    u += 0x7fffu + ((u >> 16) & 1u);   // round-to-nearest-even
    return (unsigned short)(u >> 16);
}

// ---------------------------------------------------------------------------
// Kernel 1: project BEV points into cameras, masked-mean bilinear sample.
// Output: pixel-major featT[pix * KC + k], k = c*5 + lvl; k in [655,672) = 0.
// ---------------------------------------------------------------------------
__global__ void __launch_bounds__(128)
ipm_project(const float* __restrict__ cam_feat,
            const float* __restrict__ ego2cam,
            const int*   __restrict__ img_h_p,
            const int*   __restrict__ img_w_p,
            unsigned short* __restrict__ featT)
{
    const int p   = blockIdx.x;               // 0 .. 143999, ordered (lvl, y, x)
    const int c   = threadIdx.x;              // 0 .. 127
    const int x   = p % BEV_W;
    const int t   = p / BEV_W;
    const int y   = t % BEV_H;
    const int lvl = t / BEV_H;

    const float xv = -30.0f + x * (60.0f / 239.0f);
    const float yv = -15.0f + y * (30.0f / 119.0f);
    const float zv = -1.0f  + lvl * 0.5f;

    const float inv_w = 1.0f / (float)img_w_p[0];
    const float inv_h = 1.0f / (float)img_h_p[0];

    float acc = 0.0f;
    float cnt = 0.0f;

    for (int n = 0; n < NCAM; ++n) {
        const float* M = ego2cam + n * 16;
        const float cx = M[0]*xv + M[1]*yv + M[2]*zv  + M[3];
        const float cy = M[4]*xv + M[5]*yv + M[6]*zv  + M[7];
        const float cz = M[8]*xv + M[9]*yv + M[10]*zv + M[11];

        if (!(cz > 1e-9f)) continue;
        const float px = ((cx / cz + 1e-9f) * inv_w - 0.5f) * 2.0f;
        const float py = ((cy / cz + 1e-9f) * inv_h - 0.5f) * 2.0f;
        if (!(px > -1.0f && px < 1.0f && py > -1.0f && py < 1.0f)) continue;

        const float sx = ((px + 1.0f) * (float)IMW - 1.0f) * 0.5f;
        const float sy = ((py + 1.0f) * (float)IMH - 1.0f) * 0.5f;
        const float x0 = floorf(sx), y0 = floorf(sy);
        const float wa = (x0 + 1.0f - sx) * (y0 + 1.0f - sy);
        const float wb = (x0 + 1.0f - sx) * (sy - y0);
        const float wc = (sx - x0) * (y0 + 1.0f - sy);
        const float wd = (sx - x0) * (sy - y0);
        const int xi = (int)x0, yi = (int)y0;

        const float* img = cam_feat + ((size_t)(n * CIN + c)) * (IMH * IMW);
        auto tap = [&](int ix, int iy) -> float {
            return ((unsigned)ix < (unsigned)IMW && (unsigned)iy < (unsigned)IMH)
                 ? img[iy * IMW + ix] : 0.0f;
        };
        acc += tap(xi, yi) * wa + tap(xi, yi + 1) * wb +
               tap(xi + 1, yi) * wc + tap(xi + 1, yi + 1) * wd;
        cnt += 1.0f;
    }

    const float v = acc / (cnt == 0.0f ? 1.0f : cnt);
    const size_t row = (size_t)(y * BEV_W + x) * KC;
    featT[row + c * NLVL + lvl] = f2bf(v);

    if (c < 3) {  // coord channels 128..130
        const float cv = (c == 0) ? xv : ((c == 1) ? yv : zv);
        featT[row + (CIN + c) * NLVL + lvl] = f2bf(cv);
    }
    if (lvl == 0 && c < KC - CFEAT) {  // zero the K padding once per pixel
        featT[row + CFEAT + c] = 0;
    }
}

// ---------------------------------------------------------------------------
// Kernel 2: pack conv weights (128,655,3,3) fp32 into per-lane WMMA B-fragment
// order: (ntile, step s = r*21+kbi, lane, e) -> 16 contiguous bf16 with
// k = kbi*32 + (lane>>4)*8 + (e&7) + (e>>3)*16, column = ntile*16 + (lane&15), tap r.
// ---------------------------------------------------------------------------
__global__ void __launch_bounds__(256)
ipm_pack_w(const float* __restrict__ w, unsigned short* __restrict__ bsw)
{
    const int i = blockIdx.x * 256 + threadIdx.x;
    if (i >= 8 * NSTEP * 32 * 16) return;
    const int e    = i & 15;
    const int t1   = i >> 4;
    const int lane = t1 & 31;
    const int t2   = t1 >> 5;
    const int s    = t2 % NSTEP;
    const int nt   = t2 / NSTEP;
    const int r    = s / NKB;          // tap 0..8
    const int kbi  = s % NKB;
    const int ci   = kbi * 32 + (lane >> 4) * 8 + (e & 7) + (e >> 3) * 16;   // channel (GEMM k)
    const int col  = nt * 16 + (lane & 15);             // output channel
    const float v  = (ci < CFEAT) ? w[(size_t)col * (CFEAT * 9) + ci * 9 + r] : 0.0f;
    bsw[i] = f2bf(v);
}

// ---------------------------------------------------------------------------
// CDNA5 async copy: global -> LDS, 16 B per lane, tracked by ASYNCcnt.
// ---------------------------------------------------------------------------
__device__ __forceinline__ void async_g2l_b128(unsigned lds_addr, const void* gaddr) {
    asm volatile("global_load_async_to_lds_b128 %0, %1, off"
                 :: "v"(lds_addr), "v"(gaddr) : "memory");
}
__device__ __forceinline__ void wait_async_le6() {
    asm volatile("s_wait_asynccnt 0x6" ::: "memory");
}
__device__ __forceinline__ void wait_async_0() {
    asm volatile("s_wait_asynccnt 0x0" ::: "memory");
}

// ---------------------------------------------------------------------------
// Kernel 3: 3x3 conv as 9 tap-GEMMs, M=28800 pixels, N=128, K=672 per tap.
// Block = 8 waves sharing one N-tile; per-tap B (21 KB) is double-buffered in
// LDS via global_load_async_to_lds_b128 (6 ops/thread/tap), synced with
// s_wait_asynccnt + workgroup barriers. A fragment = two contiguous global
// b128 loads; B fragment = one 32 B LDS read per WMMA.
// ---------------------------------------------------------------------------
__global__ void __launch_bounds__(256)
ipm_conv_wmma(const unsigned short* __restrict__ featT,
              const unsigned short* __restrict__ bsw,
              const float* __restrict__ bias,
              float* __restrict__ out)
{
    __shared__ unsigned short bsm[2][STAGEB / 2];   // 2 x 24 KB

    const int tid  = threadIdx.x;
    const int wave = tid >> 5;
    const int lane = tid & 31;
    const int nt     = blockIdx.x & 7;                 // 0..7  N-tile
    const int mgroup = blockIdx.x >> 3;                // 0..224
    const int mtile  = mgroup * 8 + wave;              // 0..1799

    const int hi  = lane >> 4;
    const int l16 = lane & 15;

    // 240 % 16 == 0: a 16-pixel M-tile stays within one BEV row
    const int oy = (mtile * 16) / BEV_W;
    const int ox = (mtile * 16) % BEV_W + l16;         // this lane's x

    const char* gtap = (const char*)(bsw + (size_t)nt * NSTEP * 512);  // tap r at +r*21504
    const unsigned lds0 = (unsigned)(uintptr_t)&bsm[0][0];
    const unsigned lds1 = (unsigned)(uintptr_t)&bsm[1][0];

    // stage tap r into buffer b: 24 KB, 6 async b128 per thread
    auto stage = [&](int r, unsigned ldsbase) {
#pragma unroll
        for (int it = 0; it < 6; ++it) {
            const unsigned off = (unsigned)(tid * 16 + it * 4096);
            async_g2l_b128(ldsbase + off, gtap + (size_t)r * (TAPB * 2) + off);
        }
    };

    stage(0, lds0);

    v8f acc = {};

    for (int r = 0; r < 9; ++r) {
        if (r < 8) stage(r + 1, (r & 1) ? lds0 : lds1);   // prefetch next tap
        if (r < 8) wait_async_le6(); else wait_async_0(); // current tap landed
        __syncthreads();                                  // block-wide visibility

        const int ky = r / 3, kx = r % 3;
        const int iy = oy + ky - 1;
        const int ix = ox + kx - 1;
        const bool aok = ((unsigned)iy < (unsigned)BEV_H) &
                         ((unsigned)ix < (unsigned)BEV_W);
        const unsigned short* arow =
            featT + (size_t)((aok ? iy : 0) * BEV_W + (aok ? ix : 0)) * KC + hi * 8;
        __builtin_prefetch(arow + KC, 0, 0);              // neighbor row stream
        const unsigned short* bbuf = &bsm[r & 1][lane * 16];

#pragma unroll 7
        for (int kbi = 0; kbi < NKB; ++kbi) {
            v8u c0 = {}, c1 = {};
            if (aok) {
                c0 = *(const v8u*)(arow + kbi * 32);        // k = kb + hi*8 ..
                c1 = *(const v8u*)(arow + kbi * 32 + 16);   // k = kb+16+hi*8 ..
            }
            const v16u au = __builtin_shufflevector(c0, c1,
                0, 1, 2, 3, 4, 5, 6, 7, 8, 9, 10, 11, 12, 13, 14, 15);
            const v16u bu = *(const v16u*)(bbuf + (size_t)kbi * 512);
            const v16bf a = __builtin_bit_cast(v16bf, au);
            const v16bf b = __builtin_bit_cast(v16bf, bu);
            acc = __builtin_amdgcn_wmma_f32_16x16x32_bf16(
                      false, a, false, b, (short)0, acc, false, false);
        }
        __syncthreads();   // everyone done reading buf before it is re-staged
    }

    // f32 C/D 16x16 layout: VGPR j -> M = hi*8 + j, N = lane & 15
    const int oc = nt * 16 + l16;
    const float bo = bias[oc];
#pragma unroll
    for (int j = 0; j < 8; ++j) {
        const int m = mtile * 16 + hi * 8 + j;
        out[(size_t)oc * NPIX + m] = acc[j] + bo;
    }
}

// ---------------------------------------------------------------------------
extern "C" void kernel_launch(void* const* d_in, const int* in_sizes, int n_in,
                              void* d_out, int out_size, void* d_ws, size_t ws_size,
                              hipStream_t stream)
{
    const float* cam_feat = (const float*)d_in[0];   // (6,128,64,176)
    const float* ego2cam  = (const float*)d_in[1];   // (1,6,4,4)
    const float* conv_w   = (const float*)d_in[2];   // (128,655,3,3)
    const float* conv_b   = (const float*)d_in[3];   // (128,)
    const int*   img_h    = (const int*)d_in[4];     // scalar 512
    const int*   img_w    = (const int*)d_in[5];     // scalar 1408

    unsigned short* featT = (unsigned short*)d_ws;               // 28800*672 bf16
    unsigned short* bsw   = featT + (size_t)NPIX * KC;           // 8*189*512 bf16 (+pad)
    float* out = (float*)d_out;                                  // (1,128,120,240)

    ipm_project<<<NLVL * NPIX, 128, 0, stream>>>(cam_feat, ego2cam, img_h, img_w, featT);
    ipm_pack_w<<<(8 * NSTEP * 32 * 16 + 255) / 256, 256, 0, stream>>>(conv_w, bsw);
    // 225 M-groups x 8 N-tiles; 8 waves/block, one 16x16 output tile per wave
    ipm_conv_wmma<<<225 * 8, 256, 0, stream>>>(featT, bsw, conv_b, out);
}
